// AdaFaceLoss_23630910062917
// MI455X (gfx1250) — matrix operand order repaired and checked by reading, not compiled, not run on hardware
//
#include <hip/hip_runtime.h>
#include <cstdint>

// ---------------- problem constants ----------------
#define NB      512          // batch
#define DIM     512          // embedding dim
#define NCLS    100000       // classes
#define SCALE_F 30.0f
#define MARGIN_F 0.4f
#define H_F     0.333f
#define CLS_PER_WG 32
#define NWG     (NCLS / CLS_PER_WG)   // 3125

// ---------------- WMMA types ----------------
typedef __attribute__((ext_vector_type(16))) __bf16 v16bf;
typedef __attribute__((ext_vector_type(8)))  float  v8f;
typedef __attribute__((ext_vector_type(4)))  float  vf4;

union Frag {
    vf4   f4[2];
    v16bf v;
};

// float -> bf16 bits, round-to-nearest-even
__device__ __forceinline__ unsigned short f2bf(float f) {
    unsigned int u = __float_as_uint(f);
    u += 0x7FFFu + ((u >> 16) & 1u);
    return (unsigned short)(u >> 16);
}
__device__ __forceinline__ unsigned int pack2(float a, float b) {
    return (unsigned int)f2bf(a) | ((unsigned int)f2bf(b) << 16);
}

// =====================================================================
// Kernel 1: per-row embedding norm + bf16 unit embeddings
// grid 512 x 128 threads
// =====================================================================
__global__ void k_norm_unit(const float* __restrict__ emb,
                            unsigned short* __restrict__ emb_unit,
                            float* __restrict__ norms) {
    const int row = blockIdx.x;
    const int t   = threadIdx.x;           // 0..127
    const float4 v = ((const float4*)(emb + (size_t)row * DIM))[t];
    float ss = v.x * v.x + v.y * v.y + v.z * v.z + v.w * v.w;
    #pragma unroll
    for (int m = 16; m >= 1; m >>= 1) ss += __shfl_xor(ss, m, 32);
    __shared__ float wsum[4];
    __shared__ float nsh;
    if ((t & 31) == 0) wsum[t >> 5] = ss;
    __syncthreads();
    if (t == 0) {
        float s = wsum[0] + wsum[1] + wsum[2] + wsum[3];
        float n = sqrtf(s);
        norms[row] = n;
        nsh = n;
    }
    __syncthreads();
    const float inv = 1.0f / nsh;
    uint2 p;
    p.x = pack2(v.x * inv, v.y * inv);
    p.y = pack2(v.z * inv, v.w * inv);
    ((uint2*)(emb_unit + (size_t)row * DIM))[t] = p;
}

// =====================================================================
// Kernel 2: batch mean / unbiased std of norms -> adaptive margins
// grid 1 x 512 threads
// =====================================================================
__global__ void k_margin(const float* __restrict__ norms,
                         float* __restrict__ margins) {
    const int t = threadIdx.x;            // 0..511
    const float x = norms[t];
    __shared__ float lds[16];
    __shared__ float sh_mean, sh_std;

    float s = x;
    #pragma unroll
    for (int m = 16; m >= 1; m >>= 1) s += __shfl_xor(s, m, 32);
    if ((t & 31) == 0) lds[t >> 5] = s;
    __syncthreads();
    if (t == 0) {
        float tot = 0.f;
        for (int i = 0; i < 16; ++i) tot += lds[i];
        sh_mean = tot / (float)NB;
    }
    __syncthreads();
    const float mean = sh_mean;
    float d  = x - mean;
    float s2 = d * d;
    #pragma unroll
    for (int m = 16; m >= 1; m >>= 1) s2 += __shfl_xor(s2, m, 32);
    if ((t & 31) == 0) lds[t >> 5] = s2;
    __syncthreads();
    if (t == 0) {
        float tot = 0.f;
        for (int i = 0; i < 16; ++i) tot += lds[i];
        sh_std = sqrtf(tot / (float)(NB - 1));   // ddof = 1
    }
    __syncthreads();
    float ms = (x - mean) / (sh_std + H_F);
    ms = fminf(1.0f, fmaxf(-1.0f, ms));
    margins[t] = MARGIN_F * (1.0f + ms);
}

// =====================================================================
// Kernel 3: cos_t[row] = emb[row].w[label] / (|emb[row]| |w[label]|)
// grid 512 x 128 threads
// =====================================================================
__global__ void k_cost(const float* __restrict__ emb,
                       const int* __restrict__ labels,
                       const float* __restrict__ weight,
                       const float* __restrict__ norms,
                       float* __restrict__ cos_t) {
    const int row = blockIdx.x;
    const int t   = threadIdx.x;
    const int lbl = labels[row];
    const float4 w = ((const float4*)(weight + (size_t)lbl * DIM))[t];
    const float4 e = ((const float4*)(emb + (size_t)row * DIM))[t];
    float dt  = w.x * e.x + w.y * e.y + w.z * e.z + w.w * e.w;
    float wsq = w.x * w.x + w.y * w.y + w.z * w.z + w.w * w.w;
    #pragma unroll
    for (int m = 16; m >= 1; m >>= 1) {
        dt  += __shfl_xor(dt,  m, 32);
        wsq += __shfl_xor(wsq, m, 32);
    }
    __shared__ float ldt[4], lwq[4];
    if ((t & 31) == 0) { ldt[t >> 5] = dt; lwq[t >> 5] = wsq; }
    __syncthreads();
    if (t == 0) {
        float sd = ldt[0] + ldt[1] + ldt[2] + ldt[3];
        float sw = lwq[0] + lwq[1] + lwq[2] + lwq[3];
        cos_t[row] = sd / (norms[row] * sqrtf(sw));
    }
}

// =====================================================================
// Kernel 4 (main): fused normalize-W + bf16 WMMA GEMM + exp-rowsum
// grid 3125 x 256 threads (8 waves). Each block: 32 classes x full batch.
// Each wave: 4 M-tiles x 2 N-tiles of 16x16, K = 512 in steps of 32.
// =====================================================================
__global__ __launch_bounds__(256) void k_gemm(
        const float* __restrict__ weight,
        const unsigned short* __restrict__ emb_unit,
        float* __restrict__ partials) {
    __shared__ unsigned short wtile[CLS_PER_WG][DIM + 8];  // +8 pad: conflict-free B reads
    __shared__ float rowsum[2][NB];

    const int t  = threadIdx.x;
    const int c0 = blockIdx.x * CLS_PER_WG;

    // ---- stage 32 weight rows: load fp32, row-normalize, store bf16 to LDS ----
    {
        const int r  = t >> 3;        // class row 0..31 (8 threads per row)
        const int c8 = t & 7;
        const float4* wrow = (const float4*)(weight + (size_t)(c0 + r) * DIM);
        float4 v[16];
        float ss = 0.f;
        #pragma unroll
        for (int i = 0; i < 16; ++i) {
            v[i] = wrow[c8 + 8 * i];
            ss += v[i].x * v[i].x + v[i].y * v[i].y + v[i].z * v[i].z + v[i].w * v[i].w;
        }
        #pragma unroll
        for (int m = 4; m >= 1; m >>= 1) ss += __shfl_xor(ss, m, 8);
        const float inv = 1.0f / sqrtf(ss);
        #pragma unroll
        for (int i = 0; i < 16; ++i) {
            uint2 p;
            p.x = pack2(v[i].x * inv, v[i].y * inv);
            p.y = pack2(v[i].z * inv, v[i].w * inv);
            *(uint2*)&wtile[r][(c8 + 8 * i) * 4] = p;
        }
    }
    __syncthreads();

    const int lane = t & 31;
    const int wv   = t >> 5;         // 0..7
    const int half = lane >> 4;      // 0/1
    const int l15  = lane & 15;

    // B fragments: lane holds column n = l15 of N-tile, K chunks at kb + half*8 and kb + half*8 + 16
    const unsigned short* bp0 = &wtile[l15][half * 8];
    const unsigned short* bp1 = &wtile[16 + l15][half * 8];
    // A fragments: lane holds row m, K chunks at kb + half*8 and kb + half*8 + 16
    const unsigned short* ap[4];
    #pragma unroll
    for (int i = 0; i < 4; ++i)
        ap[i] = emb_unit + (size_t)((wv + 8 * i) * 16 + l15) * DIM + half * 8;

    v8f acc0[4], acc1[4];
    const v8f vzero = {0.f, 0.f, 0.f, 0.f, 0.f, 0.f, 0.f, 0.f};
    #pragma unroll
    for (int i = 0; i < 4; ++i) { acc0[i] = vzero; acc1[i] = vzero; }

    #pragma unroll 2
    for (int kb = 0; kb < DIM; kb += 32) {
        Frag b0, b1;
        b0.f4[0] = *(const vf4*)(bp0 + kb);
        b0.f4[1] = *(const vf4*)(bp0 + kb + 16);
        b1.f4[0] = *(const vf4*)(bp1 + kb);
        b1.f4[1] = *(const vf4*)(bp1 + kb + 16);
        #pragma unroll
        for (int i = 0; i < 4; ++i) {
            Frag a;
            a.f4[0] = *(const vf4*)(ap[i] + kb);
            a.f4[1] = *(const vf4*)(ap[i] + kb + 16);
            acc0[i] = __builtin_amdgcn_wmma_f32_16x16x32_bf16(
                          false, a.v, false, b0.v, (short)0, acc0[i], false, false);
            acc1[i] = __builtin_amdgcn_wmma_f32_16x16x32_bf16(
                          false, a.v, false, b1.v, (short)0, acc1[i], false, false);
        }
    }

    // ---- epilogue: exp(30*(cos-1)), per-row sum over the 16 classes of each tile ----
    #pragma unroll
    for (int i = 0; i < 4; ++i) {
        #pragma unroll
        for (int r = 0; r < 8; ++r) {
            float e0 = __expf((acc0[i][r] - 1.0f) * SCALE_F);
            float e1 = __expf((acc1[i][r] - 1.0f) * SCALE_F);
            #pragma unroll
            for (int m = 8; m >= 1; m >>= 1) {
                e0 += __shfl_xor(e0, m, 16);
                e1 += __shfl_xor(e1, m, 16);
            }
            // lanes 0-15 reduced rows M=r ; lanes 16-31 reduced rows M=r+8
            if (l15 == r) {
                const int mrow = (wv + 8 * i) * 16 + r + half * 8;
                rowsum[0][mrow] = e0;
                rowsum[1][mrow] = e1;
            }
        }
    }
    __syncthreads();

    float* pout = partials + (size_t)blockIdx.x * NB;
    pout[t]       = rowsum[0][t]       + rowsum[1][t];
    pout[t + 256] = rowsum[0][t + 256] + rowsum[1][t + 256];
}

// =====================================================================
// Kernel 5: reduce partials[3125][512] -> rowsumG[512] (coalesced, deterministic)
// grid 32 x 256 threads
// =====================================================================
__global__ void k_reduce(const float* __restrict__ partials,
                         float* __restrict__ rowsumG) {
    __shared__ float lds[16][17];
    const int t  = threadIdx.x;
    const int ty = t & 15;        // row within group (consecutive threads -> coalesced)
    const int tx = t >> 4;        // chunk of workgroups
    const int m  = blockIdx.x * 16 + ty;
    float s = 0.f;
    for (int i = tx; i < NWG; i += 16) s += partials[(size_t)i * NB + m];
    lds[tx][ty] = s;
    __syncthreads();
    if (tx == 0) {
        float tot = 0.f;
        for (int i = 0; i < 16; ++i) tot += lds[i][ty];
        rowsumG[m] = tot;
    }
}

// =====================================================================
// Kernel 6: margin correction + logsumexp + mean loss
// grid 1 x 512 threads
// =====================================================================
__global__ void k_final(const float* __restrict__ rowsumG,
                        const float* __restrict__ cos_t,
                        const float* __restrict__ margins,
                        float* __restrict__ out) {
    const int t = threadIdx.x;
    const float S  = rowsumG[t];
    const float ct = fminf(1.0f, fmaxf(-1.0f, cos_t[t]));
    const float am = margins[t];
    const float ctm = cosf(acosf(ct) + am);
    // replace label term, lse with fixed max = SCALE
    const float Sp = S - expf((ct - 1.0f) * SCALE_F) + expf((ctm - 1.0f) * SCALE_F);
    float li = (logf(Sp) + SCALE_F) - SCALE_F * ctm;
    __shared__ float lds[16];
    #pragma unroll
    for (int m = 16; m >= 1; m >>= 1) li += __shfl_xor(li, m, 32);
    if ((t & 31) == 0) lds[t >> 5] = li;
    __syncthreads();
    if (t == 0) {
        float tot = 0.f;
        for (int i = 0; i < 16; ++i) tot += lds[i];
        out[0] = tot / (float)NB;
    }
}

// =====================================================================
extern "C" void kernel_launch(void* const* d_in, const int* in_sizes, int n_in,
                              void* d_out, int out_size, void* d_ws, size_t ws_size,
                              hipStream_t stream) {
    const float* emb    = (const float*)d_in[0];   // [512,512] fp32
    const int*   labels = (const int*)d_in[1];     // [512] int
    const float* weight = (const float*)d_in[2];   // [100000,512] fp32

    char* ws = (char*)d_ws;
    unsigned short* emb_unit = (unsigned short*)(ws);                  // 512*512*2 = 524288 B
    float* norms    = (float*)(ws + 524288);                           // 2048 B
    float* margins  = (float*)(ws + 526336);                           // 2048 B
    float* cos_t    = (float*)(ws + 528384);                           // 2048 B
    float* rowsumG  = (float*)(ws + 530432);                           // 2048 B
    float* partials = (float*)(ws + 532480);                           // 3125*512*4 = 6.4 MB

    k_norm_unit<<<NB, 128, 0, stream>>>(emb, emb_unit, norms);
    k_margin<<<1, NB, 0, stream>>>(norms, margins);
    k_cost<<<NB, 128, 0, stream>>>(emb, labels, weight, norms, cos_t);
    k_gemm<<<NWG, 256, 0, stream>>>(weight, emb_unit, partials);
    k_reduce<<<NB / 16, 256, 0, stream>>>(partials, rowsumG);
    k_final<<<1, NB, 0, stream>>>(rowsumG, cos_t, margins, (float*)d_out);
}
